// LatencyEncoder_32701880992463
// MI455X (gfx1250) — matrix operand order repaired and checked by reading, not compiled, not run on hardware
//
#include <hip/hip_runtime.h>
#include <hip/hip_bf16.h>

typedef __attribute__((ext_vector_type(16))) _Float16 v16h;
typedef __attribute__((ext_vector_type(8)))  float    v8f;

namespace {
constexpr int kT      = 64;
constexpr int kNObs   = 256;
constexpr int kH      = 1024;
constexpr int kBatch  = 4096;
constexpr int kNT     = 4;           // 4 WMMA N-tiles (64 h columns) per wave
constexpr int kWaves  = 4;           // waves per block (128 threads)
}

__global__ __launch_bounds__(128, 2)
void snn_latency_fused(const float* __restrict__ x,       // [4096, 256]
                       const float* __restrict__ mem_in,  // [4096, 1024]
                       const float* __restrict__ W,       // [1024, 256]
                       const float* __restrict__ bias,    // [1024]
                       const float* __restrict__ beta_p,  // [1]
                       float* __restrict__ out,           // [4096, 1024]
                       float* __restrict__ mem_out)       // [4096, 1024]
{
  const int lane = threadIdx.x & 31;
  const int wave = threadIdx.x >> 5;
  const int w    = blockIdx.x * kWaves + wave;

  const int h_grp    = w & 15;       // 1024 / 64 = 16 column groups
  const int row_tile = w >> 4;       // 4096 / 16 = 256 row tiles
  const int row0 = row_tile * 16;
  const int h0   = h_grp * 64;

  const int nl  = lane & 15;         // A row (m) / B,C column (n)
  const int grp = lane >> 4;         // lane half (0 or 1)
  const int kbA = grp * 16;          // A: K offset (0 or 16) inside 32-chunk
  const int kbB = grp * 16;          // B: K offset (0 or 16) inside 32-chunk

  // ---------------- GEMM phase: acc[nt] = x[row0:+16] * W^T[:, h0+16*nt:+16]
  v8f acc[kNT];
#pragma unroll
  for (int nt = 0; nt < kNT; ++nt)
#pragma unroll
    for (int r = 0; r < 8; ++r) acc[nt][r] = 0.0f;

  const float* xrow = x + (row0 + nl) * kNObs;

  for (int k0 = 0; k0 < kNObs; k0 += 32) {
    // A fragment (contiguous k order, k = kbA + s, same as B):
    //   v16h[0..7]  = x[m][k0 + kbA + 0..7]
    //   v16h[8..15] = x[m][k0 + kbA + 8..15]
    const float4* pa0 = (const float4*)(xrow + k0 + kbA);
    const float4* pa1 = (const float4*)(xrow + k0 + kbA + 8);
    float4 f0 = pa0[0], f1 = pa0[1];
    float4 f2 = pa1[0], f3 = pa1[1];
    v16h a;
    a[0]  = (_Float16)f0.x; a[1]  = (_Float16)f0.y; a[2]  = (_Float16)f0.z; a[3]  = (_Float16)f0.w;
    a[4]  = (_Float16)f1.x; a[5]  = (_Float16)f1.y; a[6]  = (_Float16)f1.z; a[7]  = (_Float16)f1.w;
    a[8]  = (_Float16)f2.x; a[9]  = (_Float16)f2.y; a[10] = (_Float16)f2.z; a[11] = (_Float16)f2.w;
    a[12] = (_Float16)f3.x; a[13] = (_Float16)f3.y; a[14] = (_Float16)f3.z; a[15] = (_Float16)f3.w;

#pragma unroll
    for (int nt = 0; nt < kNT; ++nt) {
      // B fragment (contiguous k order, same as A): lane holds n = nl,
      //   K = kbB + j for j = 0..15  -> 16 contiguous floats of W row (W^T col)
      const float* wrow = W + (h0 + nt * 16 + nl) * kNObs + k0 + kbB;
      const float4* pb = (const float4*)wrow;
      float4 g0 = pb[0], g1 = pb[1], g2 = pb[2], g3 = pb[3];
      v16h bm;
      bm[0]  = (_Float16)g0.x; bm[1]  = (_Float16)g0.y; bm[2]  = (_Float16)g0.z; bm[3]  = (_Float16)g0.w;
      bm[4]  = (_Float16)g1.x; bm[5]  = (_Float16)g1.y; bm[6]  = (_Float16)g1.z; bm[7]  = (_Float16)g1.w;
      bm[8]  = (_Float16)g2.x; bm[9]  = (_Float16)g2.y; bm[10] = (_Float16)g2.z; bm[11] = (_Float16)g2.w;
      bm[12] = (_Float16)g3.x; bm[13] = (_Float16)g3.y; bm[14] = (_Float16)g3.z; bm[15] = (_Float16)g3.w;

      acc[nt] = __builtin_amdgcn_wmma_f32_16x16x32_f16(
          /*neg_a=*/false, a, /*neg_b=*/false, bm,
          /*c_mod=*/(short)0, acc[nt], /*reuse_a=*/false, /*reuse_b=*/false);
    }
  }

  // ---------------- Epilogue: bias + sigmoid, load initial membrane
  float beta = beta_p[0];
  beta = fminf(fmaxf(beta, 0.0f), 1.0f);

  // C/D 32-bit layout: VGPR r -> M = grp*8 + r, N = nl
  float cur0[kNT][8];   // sigmoid(x@W^T + b), the per-element drive
  float memv[kNT][8];
  float spks[kNT][8];
#pragma unroll
  for (int nt = 0; nt < kNT; ++nt) {
    const int h = h0 + nt * 16 + nl;
    const float bv = bias[h];
#pragma unroll
    for (int r = 0; r < 8; ++r) {
      const int row = row0 + grp * 8 + r;
      const float z = acc[nt][r] + bv;
      cur0[nt][r] = 1.0f / (1.0f + __expf(-z));
      memv[nt][r] = mem_in[row * kH + h];
      spks[nt][r] = 0.0f;
    }
  }

  // ---------------- 64-step LIF scan, all in registers
  const float decay = 0.81873075307798185867f;  // exp(-1/5)
  float tw = 1.0f;                               // exp(-t/5), t = 0
  for (int t = 0; t < kT; ++t) {
#pragma unroll
    for (int nt = 0; nt < kNT; ++nt)
#pragma unroll
      for (int r = 0; r < 8; ++r) {
        const float mo    = memv[nt][r];
        const float reset = (mo > 1.0f) ? 1.0f : 0.0f;   // from previous mem
        const float mn    = fmaf(beta, mo, cur0[nt][r] * tw) - reset;
        memv[nt][r] = mn;
        spks[nt][r] += (mn > 1.0f) ? 1.0f : 0.0f;
      }
    tw *= decay;
  }

  // ---------------- Write out = max(spk_rate, 0.05) and mem_final
#pragma unroll
  for (int nt = 0; nt < kNT; ++nt) {
    const int h = h0 + nt * 16 + nl;
#pragma unroll
    for (int r = 0; r < 8; ++r) {
      const int row = row0 + grp * 8 + r;
      const int idx = row * kH + h;
      out[idx]     = fmaxf(spks[nt][r] * (1.0f / 64.0f), 0.05f);
      mem_out[idx] = memv[nt][r];
    }
  }
}

extern "C" void kernel_launch(void* const* d_in, const int* in_sizes, int n_in,
                              void* d_out, int out_size, void* d_ws, size_t ws_size,
                              hipStream_t stream) {
  (void)in_sizes; (void)n_in; (void)d_ws; (void)ws_size; (void)out_size;
  const float* x      = (const float*)d_in[0];   // [4096, 256]
  const float* mem_in = (const float*)d_in[1];   // [4096, 1024]
  const float* W      = (const float*)d_in[2];   // [1024, 256]
  const float* bias   = (const float*)d_in[3];   // [1024]
  const float* beta   = (const float*)d_in[4];   // [1]

  float* out     = (float*)d_out;                // first output, 4096*1024
  float* mem_out = out + (size_t)kBatch * kH;    // second output, 4096*1024

  // 4096 waves total = 256 row-tiles x 16 column-groups; 4 waves per block
  const int total_waves = (kBatch / 16) * (kH / 64);
  const int blocks = total_waves / kWaves;       // 1024
  snn_latency_fused<<<blocks, kWaves * 32, 0, stream>>>(
      x, mem_in, W, bias, beta, out, mem_out);
}
